// HockeyGNN_12884901888424
// MI455X (gfx1250) — matrix operand order repaired and checked by reading, not compiled, not run on hardware
//
#include <hip/hip_runtime.h>
#include <hip/hip_bf16.h>

typedef __attribute__((ext_vector_type(16))) __bf16 v16bf;
typedef __attribute__((ext_vector_type(8)))  float  v8f;

#define NNODE 100000
#define NEDGE 1600000
#define HID   64
#define NEG_SLOPE 0.2f

// ---------- helpers: monotonic float<->uint encoding for atomicMax on f32 ----------
__device__ __forceinline__ unsigned enc_f32(float f) {
  unsigned u = __float_as_uint(f);
  return (u & 0x80000000u) ? ~u : (u | 0x80000000u);
}
__device__ __forceinline__ float dec_f32(unsigned e) {
  return (e & 0x80000000u) ? __uint_as_float(e & 0x7FFFFFFFu) : __uint_as_float(~e);
}

// ---------- CDNA5 async global->LDS copy (16B per lane), ASYNCcnt-tracked ----------
__device__ __forceinline__ void async_g2l_b128(unsigned lds_off, const void* gaddr) {
  asm volatile("global_load_async_to_lds_b128 %0, %1, off"
               :: "v"(lds_off), "v"((unsigned long long)(uintptr_t)gaddr)
               : "memory");
}
__device__ __forceinline__ void wait_async0() {
  asm volatile("s_wait_asynccnt 0" ::: "memory");
}

// B-fragment swizzle slot for a K x 64 weight matrix (bf16 WMMA 16x16x32 B layout):
// lane (half*16+m) holds K = kk*32 + half*8 + i (i<8), kk*32 + 16 + half*8 + (i-8) (i>=8) at column n = t*16 + m.
__device__ __forceinline__ int bslot(int k, int n) {
  int kk = k >> 5, r = k & 31;
  int half = (r >> 3) & 1, i = (r & 7) + ((r >> 4) << 3);
  int t = n >> 4, m = n & 15;
  return (((kk * 4 + t) * 32) + (half * 16 + m)) * 16 + i;
}

// ======================================================================
// GEMM: Y[M,64] = act_in(X[M,64]) @ W[64,64] (+bias, +relu_out optional)
// W is staged once per block into LDS in WMMA-B fragment order (bf16),
// so each B fragment is one aligned 32B LDS vector read.
// One wave = 16 rows x 64 cols, 8 v_wmma per wave.
// ======================================================================
__global__ void gemm64_wmma(const float* __restrict__ X, const float* __restrict__ W,
                            const float* __restrict__ bias, float* __restrict__ Y,
                            int M, int relu_in, int relu_out, int use_bias)
{
  __shared__ __align__(16) __bf16 wsw[2 * 4 * 32 * 16];   // 8 KB, frag-swizzled

  int tid = threadIdx.x;
  // ---- cooperative stage: coalesced global reads, swizzled bf16 LDS writes ----
  {
    int n  = tid & 63;
    int k0 = tid >> 6;                       // 0..3
    #pragma unroll
    for (int it = 0; it < 16; ++it) {
      int k = k0 + it * 4;                   // 0..63
      wsw[bslot(k, n)] = (__bf16)W[k * HID + n];
    }
  }
  __syncthreads();

  int wave = tid >> 5;
  int lane = tid & 31;
  int row0 = blockIdx.x * 128 + wave * 16;
  if (row0 >= M) return;                     // uniform per wave; no barriers below
  int m    = lane & 15;
  int half = lane >> 4;

  const float* xr = X + (size_t)(row0 + m) * HID;
  const v16bf* fr = reinterpret_cast<const v16bf*>(wsw);

  v8f c[4];
  #pragma unroll
  for (int t = 0; t < 4; ++t)
    #pragma unroll
    for (int i = 0; i < 8; ++i) c[t][i] = 0.0f;

  #pragma unroll
  for (int kk = 0; kk < 2; ++kk) {
    // A frag (16x32 bf16): lanes 0-15 carry K in {0..7,16..23}, lanes 16-31 {8..15,24..31}
    v16bf a;
    int ka = kk * 32 + half * 8;
    #pragma unroll
    for (int i = 0; i < 8; ++i) {
      float v0 = xr[ka + i];
      float v1 = xr[ka + 16 + i];
      if (relu_in) { v0 = v0 > 0.0f ? v0 : 0.0f; v1 = v1 > 0.0f ? v1 : 0.0f; }
      a[i]     = (__bf16)v0;
      a[i + 8] = (__bf16)v1;
    }
    #pragma unroll
    for (int t = 0; t < 4; ++t) {
      v16bf b = fr[(kk * 4 + t) * 32 + lane];
      c[t] = __builtin_amdgcn_wmma_f32_16x16x32_bf16(false, a, false, b,
                                                     (short)0, c[t], false, false);
    }
  }

  // C layout: VGPR v -> row v (lanes 0-15) / row v+8 (lanes 16-31), col = t*16 + m
  #pragma unroll
  for (int t = 0; t < 4; ++t) {
    int col  = t * 16 + m;
    float bv = use_bias ? bias[col] : 0.0f;
    #pragma unroll
    for (int v = 0; v < 8; ++v) {
      int row   = row0 + v + half * 8;
      float val = c[t][v] + bv;
      if (relu_out) val = val > 0.0f ? val : 0.0f;
      Y[(size_t)row * HID + col] = val;
    }
  }
}

// ---------- per-node attention logits e_src/e_dst ----------
__global__ void attn_logits(const float* __restrict__ h, const float* __restrict__ a_s,
                            const float* __restrict__ a_d, float* __restrict__ es,
                            float* __restrict__ ed, int n)
{
  int i = blockIdx.x * blockDim.x + threadIdx.x;
  if (i >= n) return;
  const float* hr = h + (size_t)i * HID;
  float s = 0.0f, d = 0.0f;
  #pragma unroll
  for (int c = 0; c < HID; ++c) { float v = hr[c]; s += v * a_s[c]; d += v * a_d[c]; }
  es[i] = s; ed[i] = d;
}

// ---------- init: out=bias broadcast, m=-inf(enc 0), Z=0 ----------
__global__ void conv_init(float* __restrict__ out, const float* __restrict__ bias,
                          unsigned* __restrict__ menc, float* __restrict__ Z, int n)
{
  int i = blockIdx.x * blockDim.x + threadIdx.x;
  if (i >= n * HID) return;
  out[i] = bias[i & (HID - 1)];
  if ((i & (HID - 1)) == 0) { int node = i >> 6; menc[node] = 0u; Z[node] = 0.0f; }
}

// ---------- pass 1: segment max over leaky_relu(e_src[s]+e_dst[d]) ----------
__global__ void attn_max(const int* __restrict__ src, const int* __restrict__ dst,
                         const float* __restrict__ es, const float* __restrict__ ed,
                         unsigned* __restrict__ menc, int E, int n)
{
  int i = blockIdx.x * blockDim.x + threadIdx.x;
  if (i >= E + n) return;
  int s, d;
  if (i < E) { s = src[i]; d = dst[i]; } else { s = d = i - E; }   // self loops appended
  float a = es[s] + ed[d];
  a = a > 0.0f ? a : NEG_SLOPE * a;
  atomicMax(menc + d, enc_f32(a));
}

// ---------- pass 2: ex = exp(alpha-m), Z = segment_sum(ex) ----------
__global__ void attn_expsum(const int* __restrict__ src, const int* __restrict__ dst,
                            const float* __restrict__ es, const float* __restrict__ ed,
                            const unsigned* __restrict__ menc, float* __restrict__ Z,
                            float* __restrict__ ex, int E, int n)
{
  int i = blockIdx.x * blockDim.x + threadIdx.x;
  if (i >= E + n) return;
  int s, d;
  if (i < E) { s = src[i]; d = dst[i]; } else { s = d = i - E; }
  float a = es[s] + ed[d];
  a = a > 0.0f ? a : NEG_SLOPE * a;
  float e = __expf(a - dec_f32(menc[d]));
  ex[i] = e;
  atomicAdd(Z + d, e);
}

// ---------- pass 3: out[d] += (ex/Z[d]) * h[s]  (one wave per edge, coalesced) ----------
__global__ void attn_aggregate(const int* __restrict__ src, const int* __restrict__ dst,
                               const float* __restrict__ h, const float* __restrict__ ex,
                               const float* __restrict__ Z, float* __restrict__ out,
                               int E, int n)
{
  int wave = threadIdx.x >> 5;
  int lane = threadIdx.x & 31;
  int i = blockIdx.x * 8 + wave;
  if (i >= E + n) return;
  int s, d;
  if (i < E) { s = src[i]; d = dst[i]; } else { s = d = i - E; }
  float w = ex[i] / (Z[d] + 1e-16f);
  const float* hs = h + (size_t)s * HID;
  float* od = out + (size_t)d * HID;
  atomicAdd(od + lane,      w * hs[lane]);
  atomicAdd(od + lane + 32, w * hs[lane + 32]);
}

// ---------- small second MLP layer: out[M,J] = hid[M,64] @ W2[64,J] + b2 ----------
__global__ void mlp_out(const float* __restrict__ hid, const float* __restrict__ W2,
                        const float* __restrict__ b2, float* __restrict__ out, int M, int J)
{
  int tid = blockIdx.x * blockDim.x + threadIdx.x;
  if (tid >= M * J) return;
  int nrow = tid / J, j = tid - nrow * J;
  float acc = b2[j];
  const float* hr = hid + (size_t)nrow * HID;
  #pragma unroll
  for (int c = 0; c < HID; ++c) acc += hr[c] * W2[c * J + j];
  out[tid] = acc;
}

// ======================================================================
// Edge MLP: out[E,4] = relu(concat(ph[src],ph[dst]) @ W1[128,64] + b1) @ W2[64,4] + b2
// W1 (32 KB f32) is async-copied to LDS (global_load_async_to_lds_b128),
// then converted once into WMMA-B fragment order (bf16, 16 KB). Per 16-edge
// tile: 16 v_wmma, B frags from LDS vector reads; second layer contracted
// in-register with half-wave shfl_xor reduction.
// ======================================================================
__global__ void edge_mlp_wmma(const float* __restrict__ ph, const int* __restrict__ src,
                              const int* __restrict__ dst, const float* __restrict__ W1,
                              const float* __restrict__ b1, const float* __restrict__ W2,
                              const float* __restrict__ b2, float* __restrict__ out,
                              int ntiles)
{
  __shared__ __align__(16) float  w1raw[128 * HID];          // 32 KB
  __shared__ __align__(16) __bf16 w1sw[4 * 4 * 32 * 16];     // 16 KB, frag-swizzled

  int tid = threadIdx.x;

  // ---- stage 1: async bulk copy W1 into LDS (no VGPR round-trip) ----
  {
    unsigned lds_base = (unsigned)(uintptr_t)&w1raw[0];
    #pragma unroll
    for (int it = 0; it < 8; ++it) {
      int v = tid + it * 256;                 // 2048 x 16B = 32 KB
      async_g2l_b128(lds_base + v * 16, (const char*)W1 + (size_t)v * 16);
    }
    wait_async0();
  }
  __syncthreads();

  // ---- stage 2: convert f32 -> bf16 into WMMA-B fragment order ----
  {
    int n  = tid & 63;
    int k0 = tid >> 6;
    #pragma unroll
    for (int it = 0; it < 32; ++it) {
      int k = k0 + it * 4;                    // 0..127
      w1sw[bslot(k, n)] = (__bf16)w1raw[k * HID + n];
    }
  }
  __syncthreads();

  int wave = tid >> 5;
  int lane = tid & 31;
  int m    = lane & 15;
  int half = lane >> 4;
  const v16bf* fr = reinterpret_cast<const v16bf*>(w1sw);

  // Hoisted second-layer weights / biases (lane-invariant across tiles)
  float w2l[4][4], b1l[4], b2l[4];
  #pragma unroll
  for (int t = 0; t < 4; ++t) {
    int col = t * 16 + m;
    b1l[t] = b1[col];
    #pragma unroll
    for (int j = 0; j < 4; ++j) w2l[t][j] = W2[col * 4 + j];
  }
  #pragma unroll
  for (int j = 0; j < 4; ++j) b2l[j] = b2[j];

  for (int tile = blockIdx.x * 8 + wave; tile < ntiles; tile += gridDim.x * 8) {
    int e = tile * 16 + m;
    int s = src[e], d = dst[e];

    v8f c[4];
    #pragma unroll
    for (int t = 0; t < 4; ++t)
      #pragma unroll
      for (int i = 0; i < 8; ++i) c[t][i] = 0.0f;

    #pragma unroll
    for (int kk = 0; kk < 4; ++kk) {
      // feature block: kk 0,1 -> ph[src], kk 2,3 -> ph[dst]; 32 features per block
      const float* base = ph + (size_t)(kk < 2 ? s : d) * HID + (kk & 1) * 32;
      v16bf a;
      int off = half * 8;
      #pragma unroll
      for (int i = 0; i < 8; ++i) {
        a[i]     = (__bf16)base[off + i];
        a[i + 8] = (__bf16)base[off + 16 + i];
      }
      #pragma unroll
      for (int t = 0; t < 4; ++t) {
        v16bf b = fr[(kk * 4 + t) * 32 + lane];
        c[t] = __builtin_amdgcn_wmma_f32_16x16x32_bf16(false, a, false, b,
                                                       (short)0, c[t], false, false);
      }
    }

    // Second layer: per-lane partials over the 4 cols this lane owns per tile t
    float p[8][4];
    #pragma unroll
    for (int v = 0; v < 8; ++v)
      #pragma unroll
      for (int j = 0; j < 4; ++j) p[v][j] = 0.0f;

    #pragma unroll
    for (int t = 0; t < 4; ++t)
      #pragma unroll
      for (int v = 0; v < 8; ++v) {
        float hv = c[t][v] + b1l[t];
        hv = hv > 0.0f ? hv : 0.0f;
        #pragma unroll
        for (int j = 0; j < 4; ++j) p[v][j] += hv * w2l[t][j];
      }

    // Reduce across the 16 lanes of each half (xor masks 1,2,4,8 never cross halves)
    #pragma unroll
    for (int v = 0; v < 8; ++v)
      #pragma unroll
      for (int j = 0; j < 4; ++j) {
        float sum = p[v][j];
        sum += __shfl_xor(sum, 1, 32);
        sum += __shfl_xor(sum, 2, 32);
        sum += __shfl_xor(sum, 4, 32);
        sum += __shfl_xor(sum, 8, 32);
        p[v][j] = sum;
      }

    // lanes 0-15 hold rows 0..7, lanes 16-31 hold rows 8..15; lane m==v writes row v
    int row_base = tile * 16 + half * 8;
    #pragma unroll
    for (int v = 0; v < 8; ++v) {
      if (m == v) {
        #pragma unroll
        for (int j = 0; j < 4; ++j)
          out[(size_t)(row_base + v) * 4 + j] = p[v][j] + b2l[j];
      }
    }
  }
}

// ======================================================================
// Host-side orchestration
// ======================================================================
extern "C" void kernel_launch(void* const* d_in, const int* in_sizes, int n_in,
                              void* d_out, int out_size, void* d_ws, size_t ws_size,
                              hipStream_t stream) {
  (void)in_sizes; (void)n_in; (void)out_size; (void)ws_size;

  // ---- inputs, in setup_inputs() dict order (params leaves in insertion order) ----
  const float* x_team   = (const float*)d_in[0];
  const float* x_player = (const float*)d_in[1];
  int idx = 2;
  const float *tW[3], *tAs[3], *tAd[3], *tB[3];
  for (int i = 0; i < 3; ++i) {
    tW[i]  = (const float*)d_in[idx++]; tAs[i] = (const float*)d_in[idx++];
    tAd[i] = (const float*)d_in[idx++]; tB[i]  = (const float*)d_in[idx++];
  }
  const float *pW[3], *pAs[3], *pAd[3], *pB[3];
  for (int i = 0; i < 3; ++i) {
    pW[i]  = (const float*)d_in[idx++]; pAs[i] = (const float*)d_in[idx++];
    pAd[i] = (const float*)d_in[idx++]; pB[i]  = (const float*)d_in[idx++];
  }
  const float* tpW1 = (const float*)d_in[idx++]; const float* tpB1 = (const float*)d_in[idx++];
  const float* tpW2 = (const float*)d_in[idx++]; const float* tpB2 = (const float*)d_in[idx++];
  const float* ppW1 = (const float*)d_in[idx++]; const float* ppB1 = (const float*)d_in[idx++];
  const float* ppW2 = (const float*)d_in[idx++]; const float* ppB2 = (const float*)d_in[idx++];
  const float* epW1 = (const float*)d_in[idx++]; const float* epB1 = (const float*)d_in[idx++];
  const float* epW2 = (const float*)d_in[idx++]; const float* epB2 = (const float*)d_in[idx++];
  const int* edge_index = (const int*)d_in[idx++];
  const int* src = edge_index;
  const int* dst = edge_index + NEDGE;

  // ---- workspace partition ----
  const size_t NH = (size_t)NNODE * HID;
  char* wp = (char*)d_ws;
  auto take = [&](size_t bytes) { void* p = wp; wp += (bytes + 255) & ~(size_t)255; return p; };
  float*    At   = (float*)take(NH * 4);
  float*    Bt   = (float*)take(NH * 4);
  float*    Ap   = (float*)take(NH * 4);
  float*    Bp   = (float*)take(NH * 4);
  float*    Hbuf = (float*)take(NH * 4);
  float*    eS   = (float*)take((size_t)NNODE * 4);
  float*    eD   = (float*)take((size_t)NNODE * 4);
  unsigned* menc = (unsigned*)take((size_t)NNODE * 4);
  float*    Zb   = (float*)take((size_t)NNODE * 4);
  float*    exb  = (float*)take((size_t)(NEDGE + NNODE) * 4);

  const int BLK = 256;
  const int g_gemm = (NNODE + 127) / 128;
  const int g_node = (NNODE + BLK - 1) / BLK;
  const int g_init = (NNODE * HID + BLK - 1) / BLK;
  const int g_edge = (NEDGE + NNODE + BLK - 1) / BLK;
  const int g_aggr = (NEDGE + NNODE + 7) / 8;

  auto run_conv = [&](const float* in, int relu_in, const float* W, const float* a_s,
                      const float* a_d, const float* bias, float* outbuf) {
    gemm64_wmma  <<<g_gemm, BLK, 0, stream>>>(in, W, nullptr, Hbuf, NNODE, relu_in, 0, 0);
    attn_logits  <<<g_node, BLK, 0, stream>>>(Hbuf, a_s, a_d, eS, eD, NNODE);
    conv_init    <<<g_init, BLK, 0, stream>>>(outbuf, bias, menc, Zb, NNODE);
    attn_max     <<<g_edge, BLK, 0, stream>>>(src, dst, eS, eD, menc, NEDGE, NNODE);
    attn_expsum  <<<g_edge, BLK, 0, stream>>>(src, dst, eS, eD, menc, Zb, exb, NEDGE, NNODE);
    attn_aggregate<<<g_aggr, BLK, 0, stream>>>(src, dst, Hbuf, exb, Zb, outbuf, NEDGE, NNODE);
  };

  // team: x -> At -> Bt -> At (final, no relu)
  run_conv(x_team, 0, tW[0], tAs[0], tAd[0], tB[0], At);
  run_conv(At,     1, tW[1], tAs[1], tAd[1], tB[1], Bt);
  run_conv(Bt,     1, tW[2], tAs[2], tAd[2], tB[2], At);
  // player: x -> Ap -> Bp -> Ap (final)
  run_conv(x_player, 0, pW[0], pAs[0], pAd[0], pB[0], Ap);
  run_conv(Ap,       1, pW[1], pAs[1], pAd[1], pB[1], Bp);
  run_conv(Bp,       1, pW[2], pAs[2], pAd[2], pB[2], Ap);

  float* out_team   = (float*)d_out;                       // [N,2]
  float* out_player = out_team + (size_t)NNODE * 2;        // [N,4]
  float* out_edge   = out_player + (size_t)NNODE * 4;      // [E,4]

  // team_pred
  gemm64_wmma<<<g_gemm, BLK, 0, stream>>>(At, tpW1, tpB1, Hbuf, NNODE, 0, 1, 1);
  mlp_out<<<(NNODE * 2 + BLK - 1) / BLK, BLK, 0, stream>>>(Hbuf, tpW2, tpB2, out_team, NNODE, 2);
  // player_pred
  gemm64_wmma<<<g_gemm, BLK, 0, stream>>>(Ap, ppW1, ppB1, Hbuf, NNODE, 0, 1, 1);
  mlp_out<<<(NNODE * 4 + BLK - 1) / BLK, BLK, 0, stream>>>(Hbuf, ppW2, ppB2, out_player, NNODE, 4);
  // edge_pred (gather + fused 2-layer MLP), E/16 = 100000 tiles
  edge_mlp_wmma<<<2048, BLK, 0, stream>>>(Ap, src, dst, epW1, epB1, epW2, epB2,
                                          out_edge, NEDGE / 16);
}
